// AttentionBlock_67508295959170
// MI455X (gfx1250) — matrix operand order repaired and checked by reading, not compiled, not run on hardware
//
#include <hip/hip_runtime.h>
#include <hip/hip_bf16.h>
#include <stdint.h>

// ---------------------------------------------------------------------------
// CDNA5 (gfx1250) wave32 WMMA attention block.
// D = A(16x32 bf16) x B(32x16 bf16) + C(16x16 f32) via v_wmma_f32_16x16x32_bf16
// ---------------------------------------------------------------------------

typedef __attribute__((ext_vector_type(16))) __bf16 v16bf;
typedef __attribute__((ext_vector_type(8)))  float  v8f;
typedef __attribute__((ext_vector_type(4)))  uint32_t u32x4;

union FragBF {            // 16 bf16 values = 8 VGPRs
  v16bf v;
  u32x4 q[2];
};

__device__ __forceinline__ uint16_t f2bf(float f) {
  uint32_t x = __builtin_bit_cast(uint32_t, f);
  uint32_t r = (x + 0x7FFFu + ((x >> 16) & 1u)) >> 16;   // round-to-nearest-even
  return (uint16_t)r;
}

__device__ __forceinline__ v8f wmma_bf16(const FragBF& a, const FragBF& b, v8f c) {
  return __builtin_amdgcn_wmma_f32_16x16x32_bf16(false, a.v, false, b.v,
                                                 (short)0, c, false, false);
}

// ---------------------------------------------------------------------------
// Kernel 1: f32 -> bf16 conversion (weights)
// ---------------------------------------------------------------------------
__global__ __launch_bounds__(256)
void cvt_bf16_kernel(const float* __restrict__ src, uint16_t* __restrict__ dst, int n) {
  for (int i = blockIdx.x * 256 + threadIdx.x; i < n; i += gridDim.x * 256)
    dst[i] = f2bf(src[i]);
}

// ---------------------------------------------------------------------------
// Kernel 2: GroupNorm over (C/32 channels x T) per (batch, group).
// Writes normalized x TRANSPOSED as bf16: xT[b][t][c]  (c contiguous).
// ---------------------------------------------------------------------------
__global__ __launch_bounds__(256)
void groupnorm_kernel(const float* __restrict__ x,
                      const float* __restrict__ gw,
                      const float* __restrict__ gb,
                      uint16_t* __restrict__ xT) {
  const int bb = blockIdx.x >> 5;          // batch
  const int g  = blockIdx.x & 31;          // group
  const int tid = threadIdx.x;
  const float* base = x + ((bb * 1024) + g * 32) * 1024;  // 32 ch x 1024 T

  float s = 0.f, s2 = 0.f;
  for (int i = tid; i < 32768; i += 256) {
    float v = base[i];
    s += v; s2 += v * v;
  }
  __shared__ float r0[256], r1[256];
  r0[tid] = s; r1[tid] = s2;
  __syncthreads();
  for (int o = 128; o > 0; o >>= 1) {
    if (tid < o) { r0[tid] += r0[tid + o]; r1[tid] += r1[tid + o]; }
    __syncthreads();
  }
  const float mu  = r0[0] * (1.0f / 32768.0f);
  const float var = r1[0] * (1.0f / 32768.0f) - mu * mu;
  const float rs  = rsqrtf(var + 1e-5f);

  for (int i = tid; i < 32768; i += 256) {
    int cl = i >> 10, t = i & 1023;
    int c = g * 32 + cl;
    float v = (base[i] - mu) * rs * gw[c] + gb[c];
    xT[((bb * 1024 + t) * 1024) + c] = f2bf(v);
  }
}

// ---------------------------------------------------------------------------
// Kernel 3: QKV GEMM.  out[b][o][t] = sum_c W[o][c]*xn[b][c][t] + bias[o]
// Wave tile 64x64: 16 WMMAs per K-step (K=1024, step 32).
// Epilogue routes o -> (head, q/k/v, c):  q,k scaled & stored as [b][h][t][64],
// v stored as [b][h][64][t] (the exact layouts the attention WMMAs need).
// ---------------------------------------------------------------------------
__global__ __launch_bounds__(128)
void qkv_gemm_kernel(const uint16_t* __restrict__ W,    // [3072][1024] bf16
                     const uint16_t* __restrict__ xT,   // [B][1024 t][1024 c]
                     const float*    __restrict__ bias, // [3072]
                     uint16_t* __restrict__ qT,         // [B][16][1024][64]
                     uint16_t* __restrict__ kT,         // [B][16][1024][64]
                     uint16_t* __restrict__ vM) {       // [B][16][64][1024]
  const int wave = threadIdx.x >> 5;
  const int lane = threadIdx.x & 31;
  const int hf = lane >> 4, lm = lane & 15;
  const int tile = blockIdx.x * 4 + wave;               // 0..3071
  const int b  = tile / 768;
  const int r  = tile % 768;
  const int o0 = (r >> 4) * 64;                         // 48 M-tiles
  const int t0 = (r & 15) * 64;                         // 16 N-tiles
  const uint16_t* xb = xT + b * (1024 * 1024);

  v8f acc[4][4];
#pragma unroll
  for (int i = 0; i < 4; ++i)
#pragma unroll
    for (int j = 0; j < 4; ++j) acc[i][j] = {};

  for (int k = 0; k < 1024; k += 32) {
    FragBF a[4], bf[4];
#pragma unroll
    for (int i = 0; i < 4; ++i) {
      const uint16_t* p = W + (o0 + i * 16 + lm) * 1024 + k + hf * 16;
      a[i].q[0] = *(const u32x4*)p;
      a[i].q[1] = *(const u32x4*)(p + 8);
      __builtin_prefetch(p + 32, 0, 1);
    }
#pragma unroll
    for (int j = 0; j < 4; ++j) {
      const uint16_t* p = xb + (t0 + j * 16 + lm) * 1024 + k + hf * 16;
      bf[j].q[0] = *(const u32x4*)p;
      bf[j].q[1] = *(const u32x4*)(p + 8);
      __builtin_prefetch(p + 32, 0, 1);
    }
#pragma unroll
    for (int i = 0; i < 4; ++i)
#pragma unroll
      for (int j = 0; j < 4; ++j)
        acc[i][j] = wmma_bf16(a[i], bf[j], acc[i][j]);
  }

  const float scl = 0.35355339059327379f;               // 64^-0.25
#pragma unroll
  for (int i = 0; i < 4; ++i) {
    const int obase = o0 + i * 16 + hf * 8;             // 8 rows, one 64-block
    const int which = (obase % 192) / 64;               // 0=q 1=k 2=v (uniform)
    const int h = obase / 192;
#pragma unroll
    for (int j = 0; j < 4; ++j) {
      const int t = t0 + j * 16 + lm;
#pragma unroll
      for (int e = 0; e < 8; ++e) {
        const int o = obase + e;
        const int c = o & 63;
        float val = acc[i][j][e] + bias[o];
        if (which == 0)
          qT[(((b * 16 + h) * 1024) + t) * 64 + c] = f2bf(val * scl);
        else if (which == 1)
          kT[(((b * 16 + h) * 1024) + t) * 64 + c] = f2bf(val * scl);
        else
          vM[(((b * 16 + h) * 64) + c) * 1024 + t] = f2bf(val);
      }
    }
  }
}

// ---------------------------------------------------------------------------
// Kernel 4: flash attention. One wave owns 16 query rows of one (b,h);
// streams s in steps of 32 with online softmax; P reshaped D-frag -> A-frag
// through per-wave LDS; accumulates a^T tile (16 t x 64 c).
// ---------------------------------------------------------------------------
__global__ __launch_bounds__(128)
void attn_kernel(const uint16_t* __restrict__ qT,
                 const uint16_t* __restrict__ kT,
                 const uint16_t* __restrict__ vM,
                 uint16_t* __restrict__ aT) {           // [B][1024 t][1024 c]
  __shared__ __align__(16) uint16_t Pb[4][16 * 32];
  const int wave = threadIdx.x >> 5;
  const int lane = threadIdx.x & 31;
  const int hf = lane >> 4, lm = lane & 15;
  const int wid = blockIdx.x * 4 + wave;                // 0..4095
  const int b  = wid >> 10;
  const int h  = (wid >> 6) & 15;
  const int t0 = (wid & 63) * 16;

  const uint16_t* qB = qT + ((b * 16 + h) * 1024) * 64;
  const uint16_t* kB = kT + ((b * 16 + h) * 1024) * 64;
  const uint16_t* vB = vM + ((b * 16 + h) * 64) * 1024;

  // Q A-fragments for the two K-steps over ch=64
  FragBF aq0, aq1;
  {
    const uint16_t* p = qB + (t0 + lm) * 64;
    aq0.q[0] = *(const u32x4*)(p + hf * 16);
    aq0.q[1] = *(const u32x4*)(p + 8 + hf * 16);
    aq1.q[0] = *(const u32x4*)(p + 32 + hf * 16);
    aq1.q[1] = *(const u32x4*)(p + 40 + hf * 16);
  }

  float m_i[8], l_i[8];
#pragma unroll
  for (int e = 0; e < 8; ++e) { m_i[e] = -__builtin_inff(); l_i[e] = 0.f; }
  v8f accv[4];
#pragma unroll
  for (int ct = 0; ct < 4; ++ct) accv[ct] = {};

  for (int s0 = 0; s0 < 1024; s0 += 32) {
    // ---- scores: S = (q*scale)^T (k*scale), two 16x16 n-tiles
    v8f S0 = {}, S1 = {};
    {
      FragBF bk;
      const uint16_t* p = kB + (s0 + lm) * 64 + hf * 16;
      bk.q[0] = *(const u32x4*)p; bk.q[1] = *(const u32x4*)(p + 8);
      S0 = wmma_bf16(aq0, bk, S0);
      bk.q[0] = *(const u32x4*)(p + 32); bk.q[1] = *(const u32x4*)(p + 40);
      S0 = wmma_bf16(aq1, bk, S0);
    }
    {
      FragBF bk;
      const uint16_t* p = kB + (s0 + 16 + lm) * 64 + hf * 16;
      bk.q[0] = *(const u32x4*)p; bk.q[1] = *(const u32x4*)(p + 8);
      S1 = wmma_bf16(aq0, bk, S1);
      bk.q[0] = *(const u32x4*)(p + 32); bk.q[1] = *(const u32x4*)(p + 40);
      S1 = wmma_bf16(aq1, bk, S1);
    }
    // ---- online softmax per row (row = hf*8+e, spread over the 16-lane half)
#pragma unroll
    for (int e = 0; e < 8; ++e) {
      float mx = fmaxf(S0[e], S1[e]);
      mx = fmaxf(mx, __shfl_xor(mx, 1, 32));
      mx = fmaxf(mx, __shfl_xor(mx, 2, 32));
      mx = fmaxf(mx, __shfl_xor(mx, 4, 32));
      mx = fmaxf(mx, __shfl_xor(mx, 8, 32));
      const float mn = fmaxf(m_i[e], mx);
      const float p0 = __expf(S0[e] - mn);
      const float p1 = __expf(S1[e] - mn);
      const float alpha = __expf(m_i[e] - mn);
      float rs = p0 + p1;
      rs += __shfl_xor(rs, 1, 32);
      rs += __shfl_xor(rs, 2, 32);
      rs += __shfl_xor(rs, 4, 32);
      rs += __shfl_xor(rs, 8, 32);
      l_i[e] = l_i[e] * alpha + rs;
      m_i[e] = mn;
      const int row = hf * 8 + e;
      Pb[wave][row * 32 + lm]      = f2bf(p0);
      Pb[wave][row * 32 + 16 + lm] = f2bf(p1);
#pragma unroll
      for (int ct = 0; ct < 4; ++ct) accv[ct][e] *= alpha;
    }
    __syncthreads();   // uniform: order LDS store -> A-frag load
    FragBF pf;
    {
      const uint16_t* p = &Pb[wave][lm * 32 + hf * 16];
      pf.q[0] = *(const u32x4*)p;
      pf.q[1] = *(const u32x4*)(p + 8);
    }
#pragma unroll
    for (int ct = 0; ct < 4; ++ct) {
      FragBF bv;
      const uint16_t* p = vB + (ct * 16 + lm) * 1024 + s0 + hf * 16;
      bv.q[0] = *(const u32x4*)p; bv.q[1] = *(const u32x4*)(p + 8);
      accv[ct] = wmma_bf16(pf, bv, accv[ct]);
    }
    __syncthreads();   // uniform: protect Pb before next iteration's store
  }

  // ---- epilogue: a^T[b][t][h*64+c] = acc / l
#pragma unroll
  for (int e = 0; e < 8; ++e) {
    const float inv = 1.0f / l_i[e];
    const int t = t0 + hf * 8 + e;
    uint16_t* po = aT + ((b * 1024 + t) * 1024) + h * 64 + lm;
#pragma unroll
    for (int ct = 0; ct < 4; ++ct)
      po[ct * 16] = f2bf(accv[ct][e] * inv);
  }
}

// ---------------------------------------------------------------------------
// Kernel 5: proj GEMM + bias + residual (f32 out).
// out[b][o][t] = x[b][o][t] + bias[o] + sum_c Wp[o][c]*a[b][c][t]
// ---------------------------------------------------------------------------
__global__ __launch_bounds__(128)
void proj_gemm_kernel(const uint16_t* __restrict__ W,   // [1024][1024] bf16
                      const uint16_t* __restrict__ aT,  // [B][1024 t][1024 c]
                      const float* __restrict__ bias,
                      const float* __restrict__ x,
                      float* __restrict__ out) {
  const int wave = threadIdx.x >> 5;
  const int lane = threadIdx.x & 31;
  const int hf = lane >> 4, lm = lane & 15;
  const int tile = blockIdx.x * 4 + wave;               // 0..1023
  const int b  = tile / 256;
  const int r  = tile % 256;
  const int o0 = (r >> 4) * 64;
  const int t0 = (r & 15) * 64;
  const uint16_t* ab = aT + b * (1024 * 1024);

  v8f acc[4][4];
#pragma unroll
  for (int i = 0; i < 4; ++i)
#pragma unroll
    for (int j = 0; j < 4; ++j) acc[i][j] = {};

  for (int k = 0; k < 1024; k += 32) {
    FragBF a[4], bf[4];
#pragma unroll
    for (int i = 0; i < 4; ++i) {
      const uint16_t* p = W + (o0 + i * 16 + lm) * 1024 + k + hf * 16;
      a[i].q[0] = *(const u32x4*)p;
      a[i].q[1] = *(const u32x4*)(p + 8);
      __builtin_prefetch(p + 32, 0, 1);
    }
#pragma unroll
    for (int j = 0; j < 4; ++j) {
      const uint16_t* p = ab + (t0 + j * 16 + lm) * 1024 + k + hf * 16;
      bf[j].q[0] = *(const u32x4*)p;
      bf[j].q[1] = *(const u32x4*)(p + 8);
      __builtin_prefetch(p + 32, 0, 1);
    }
#pragma unroll
    for (int i = 0; i < 4; ++i)
#pragma unroll
      for (int j = 0; j < 4; ++j)
        acc[i][j] = wmma_bf16(a[i], bf[j], acc[i][j]);
  }

#pragma unroll
  for (int i = 0; i < 4; ++i) {
    const int obase = o0 + i * 16 + hf * 8;
#pragma unroll
    for (int j = 0; j < 4; ++j) {
      const int t = t0 + j * 16 + lm;
#pragma unroll
      for (int e = 0; e < 8; ++e) {
        const int o = obase + e;
        const int idx = (b * 1024 + o) * 1024 + t;
        out[idx] = x[idx] + bias[o] + acc[i][j][e];
      }
    }
  }
}

// ---------------------------------------------------------------------------
// Launcher
// ---------------------------------------------------------------------------
extern "C" void kernel_launch(void* const* d_in, const int* in_sizes, int n_in,
                              void* d_out, int out_size, void* d_ws, size_t ws_size,
                              hipStream_t stream) {
  (void)in_sizes; (void)n_in; (void)out_size; (void)ws_size;
  const float* x      = (const float*)d_in[0];
  const float* norm_w = (const float*)d_in[1];
  const float* norm_b = (const float*)d_in[2];
  const float* qkv_w  = (const float*)d_in[3];
  const float* qkv_b  = (const float*)d_in[4];
  const float* proj_w = (const float*)d_in[5];
  const float* proj_b = (const float*)d_in[6];
  float* out = (float*)d_out;

  char* ws = (char*)d_ws;                       // 48 MB used, all 64B-aligned
  uint16_t* wqkv  = (uint16_t*)(ws + 0);                  // 6 MB
  uint16_t* wproj = (uint16_t*)(ws + 6291456);            // 2 MB
  uint16_t* xT    = (uint16_t*)(ws + 8388608);            // 8 MB
  uint16_t* qT    = (uint16_t*)(ws + 16777216);           // 8 MB
  uint16_t* kT    = (uint16_t*)(ws + 25165824);           // 8 MB
  uint16_t* vM    = (uint16_t*)(ws + 33554432);           // 8 MB
  uint16_t* aT    = (uint16_t*)(ws + 41943040);           // 8 MB

  cvt_bf16_kernel<<<1536, 256, 0, stream>>>(qkv_w, wqkv, 3072 * 1024);
  cvt_bf16_kernel<<<512, 256, 0, stream>>>(proj_w, wproj, 1024 * 1024);
  groupnorm_kernel<<<128, 256, 0, stream>>>(x, norm_w, norm_b, xT);
  qkv_gemm_kernel<<<768, 128, 0, stream>>>(wqkv, xT, qkv_b, qT, kT, vM);
  attn_kernel<<<1024, 128, 0, stream>>>(qT, kT, vM, aT);
  proj_gemm_kernel<<<256, 128, 0, stream>>>(wproj, aT, proj_b, x, out);
}
